// Transformer_15702400434538
// MI455X (gfx1250) — matrix operand order repaired and checked by reading, not compiled, not run on hardware
//
#include <hip/hip_runtime.h>

// ---------------- model constants (match reference) ----------------
#define DIM   2048
#define NLAY  2
#define HD    128
#define NH    16
#define NKV   4
#define FF    8192
#define VOCAB 32000
#define BATCH 2
#define SEQL  2048
#define WIN   1024
#define EPSF  1e-5f

#define BL (BATCH * SEQL)   // 4096 token rows

// ---------------- WMMA types ----------------
typedef __bf16 bf16_t;
typedef bf16_t v16bf __attribute__((ext_vector_type(16)));
typedef float  v8f   __attribute__((ext_vector_type(8)));

union Frag {
    unsigned int u[8];
    v16bf        v;
};
static_assert(sizeof(Frag) == 32, "frag size");

static __device__ __forceinline__ unsigned short to_bf16(float f) {
    unsigned int u = __float_as_uint(f);
    unsigned int r = u + 0x7FFFu + ((u >> 16) & 1u);   // round-to-nearest-even
    return (unsigned short)(r >> 16);
}

// LDS byte offset of a __shared__ object: low 32 bits of the generic
// (shared-aperture) address are the workgroup-relative LDS offset.
static __device__ __forceinline__ unsigned lds_off(const void* p) {
    return (unsigned)(unsigned long long)p;
}

// CDNA5 async global->LDS copy (ASYNCcnt-tracked, no VGPR data path).
static __device__ __forceinline__ void async_copy_b128(unsigned ldsByteOff,
                                                       const void* gaddr) {
    asm volatile("global_load_async_to_lds_b128 %0, %1, off"
                 :: "v"(ldsByteOff), "v"(gaddr)
                 : "memory");
}

// ---------------- elementwise / conversion kernels ----------------
__global__ void k_embed(const int* __restrict__ ids,
                        const float* __restrict__ emb,
                        float* __restrict__ h) {
    size_t n = (size_t)BL * DIM;
    for (size_t i = blockIdx.x * (size_t)blockDim.x + threadIdx.x; i < n;
         i += (size_t)gridDim.x * blockDim.x) {
        size_t row = i >> 11;            // DIM = 2048
        int    col = (int)(i & 2047);
        h[i] = emb[(size_t)ids[row] * DIM + col];
    }
}

__global__ void k_silu_mul(const float* __restrict__ a,
                           const float* __restrict__ b,
                           unsigned short* __restrict__ out, size_t n) {
    for (size_t i = blockIdx.x * (size_t)blockDim.x + threadIdx.x; i < n;
         i += (size_t)gridDim.x * blockDim.x) {
        float x = a[i];
        float g = x / (1.f + __expf(-x));
        out[i] = to_bf16(g * b[i]);
    }
}

// Transpose + convert: src f32 (K x N) -> dst bf16 (N x K).
// LDS-tiled 32x32 so both global streams stay coalesced (conversion is
// replayed every launch, so it must run at bandwidth).
__global__ __launch_bounds__(256) void k_transpose_bf16(
    const float* __restrict__ src, unsigned short* __restrict__ dst,
    int K, int N) {
    __shared__ float tile[32][33];
    int n0 = blockIdx.x * 32;
    int k0 = blockIdx.y * 32;
    int tx = threadIdx.x & 31;
    int ty = threadIdx.x >> 5;   // 0..7
#pragma unroll
    for (int r = 0; r < 4; ++r) {
        int kk = ty + r * 8;
        tile[kk][tx] = src[(size_t)(k0 + kk) * N + n0 + tx];
    }
    __syncthreads();
#pragma unroll
    for (int r = 0; r < 4; ++r) {
        int nn = ty + r * 8;
        dst[(size_t)(n0 + nn) * K + k0 + tx] = to_bf16(tile[tx][nn]);
    }
}

// RoPE on f32 GEMM output -> bf16, layout (B, L, nheads, HD), pair-interleaved
__global__ void k_rope(const float* __restrict__ src,
                       unsigned short* __restrict__ dst, int nheads) {
    size_t total = (size_t)BL * nheads * (HD / 2);
    const float lnTheta = __logf(10000.0f);
    for (size_t e = blockIdx.x * (size_t)blockDim.x + threadIdx.x; e < total;
         e += (size_t)gridDim.x * blockDim.x) {
        int    pr = (int)(e & (HD / 2 - 1));  // 64 pairs
        size_t t  = e >> 6;
        t /= (size_t)nheads;                  // drop head index
        int pos = (int)(t % SEQL);
        float inv = __expf(-((2.0f * pr) / (float)HD) * lnTheta);
        float ang = (float)pos * inv;
        float c = __cosf(ang), s = __sinf(ang);
        float x1 = src[2 * e], x2 = src[2 * e + 1];
        dst[2 * e]     = to_bf16(x1 * c - x2 * s);
        dst[2 * e + 1] = to_bf16(x1 * s + x2 * c);
    }
}

// RMSNorm: one block per token row, output bf16 (x * rsqrt(mean(x^2)+eps) * w)
__global__ __launch_bounds__(256) void k_rmsnorm(const float* __restrict__ x,
                                                 const float* __restrict__ w,
                                                 unsigned short* __restrict__ out) {
    __shared__ float red[256];
    size_t row = blockIdx.x;
    const float* xr = x + row * DIM;
    float ss = 0.f;
    for (int c = threadIdx.x; c < DIM; c += 256) { float v = xr[c]; ss += v * v; }
    red[threadIdx.x] = ss;
    __syncthreads();
    for (int s = 128; s > 0; s >>= 1) {
        if ((int)threadIdx.x < s) red[threadIdx.x] += red[threadIdx.x + s];
        __syncthreads();
    }
    float inv = rsqrtf(red[0] / (float)DIM + EPSF);
    for (int c = threadIdx.x; c < DIM; c += 256)
        out[row * DIM + c] = to_bf16(xr[c] * inv * w[c]);
}

// ---------------- bf16 WMMA GEMM (async double-buffered) ----------------
// C(MxN) = A(MxK) * B(KxN); A row-major, B passed PRE-TRANSPOSED as Bt (NxK).
// f32 accumulate.
// mode 0: outF = acc      mode 1: outF = res + acc      mode 2: outB = bf16(acc)
// Per K-step of 32: A tile 128x32 bf16 (8KB) and Bt tile 64x32 bf16 (4KB) are
// staged by GLOBAL_LOAD_ASYNC_TO_LDS_B128 into double buffers; each wave
// issues exactly 3 async instructions per tile, so s_wait_asynccnt 3 after
// issuing tile t+1 guarantees tile t has landed (async loads complete in
// order); a barrier then makes it visible workgroup-wide. Both A and B
// fragments are contiguous ds_load_b128 reads.
__global__ __launch_bounds__(256) void k_gemm_bf16(
    const unsigned short* __restrict__ A, const unsigned short* __restrict__ Bt,
    float* __restrict__ outF, unsigned short* __restrict__ outB,
    const float* __restrict__ res, int M, int N, int K, int mode) {
    __shared__ unsigned int As32[2 * 128 * 16];  // 2 x (128 rows x 32 bf16)
    __shared__ unsigned int Bs32[2 * 64 * 16];   // 2 x (64 cols  x 32 bf16)

    const int tid  = threadIdx.x;
    const int lane = tid & 31;
    const int wave = tid >> 5;
    const int wm   = (wave & 3) << 5;   // wave row offset within 128
    const int wn   = (wave >> 2) << 5;  // wave col offset within 64
    const int bm   = blockIdx.y * 128;
    const int bn   = blockIdx.x * 64;

    const unsigned aBase = lds_off(As32);
    const unsigned bBase = lds_off(Bs32);

    v8f acc[2][2];
#pragma unroll
    for (int i = 0; i < 2; ++i)
#pragma unroll
        for (int j = 0; j < 2; ++j)
#pragma unroll
            for (int e = 0; e < 8; ++e) acc[i][j][e] = 0.f;

    const int ln15  = lane & 15;
    const int ahalf = (lane >> 4) * 4;  // A K-offset in u32 units (0 or 4)
    const int bhalf = (lane >> 4) * 8;  // unused (B reads use K-offset 0 or 4 u32)

    // async stage of one K-tile into buffer `buf` (3 async instrs per wave)
    auto stage = [&](int k0, int buf) {
        // A: 128x32 bf16 = 512 x 16B chunks; 2 per thread
#pragma unroll
        for (int it = 0; it < 2; ++it) {
            int c16 = tid + it * 256;
            int r = c16 >> 2, cw = c16 & 3;
            async_copy_b128(aBase + (unsigned)(buf * 8192 + r * 64 + cw * 16),
                            (const void*)(A + (size_t)(bm + r) * K + k0 + cw * 8));
        }
        // Bt: 64 rows x 32 bf16 = 256 x 16B chunks; 1 per thread
        {
            int r = tid >> 2, cw = tid & 3;
            async_copy_b128(bBase + (unsigned)(buf * 4096 + r * 64 + cw * 16),
                            (const void*)(Bt + (size_t)(bn + r) * K + k0 + cw * 8));
        }
    };

    const int T = K >> 5;
    stage(0, 0);
    for (int t = 0; t < T; ++t) {
        const int cur = t & 1;
        if (t + 1 < T) {
            stage((t + 1) << 5, cur ^ 1);
            asm volatile("s_wait_asynccnt 0x3" ::: "memory");
        } else {
            asm volatile("s_wait_asynccnt 0x0" ::: "memory");
        }
        __syncthreads();

        const unsigned int* Ab = As32 + cur * 2048;
        const unsigned int* Bb = Bs32 + cur * 1024;

        Frag af[2], bf[2];
#pragma unroll
        for (int i = 0; i < 2; ++i) {
            int r = wm + i * 16 + ln15;
#pragma unroll
            for (int j = 0; j < 4; ++j) {
                af[i].u[j]     = Ab[r * 16 + ahalf + j];
                af[i].u[4 + j] = Ab[r * 16 + 8 + ahalf + j];
            }
        }
#pragma unroll
        for (int j = 0; j < 2; ++j) {
            int n = wn + j * 16 + ln15;
#pragma unroll
            for (int q = 0; q < 8; ++q) bf[j].u[q] = Bb[n * 16 + (q >> 2) * 8 + (lane >> 4) * 4 + (q & 3)];
        }
#pragma unroll
        for (int i = 0; i < 2; ++i)
#pragma unroll
            for (int j = 0; j < 2; ++j)
                acc[i][j] = __builtin_amdgcn_wmma_f32_16x16x32_bf16(
                    false, af[i].v, false, bf[j].v, (short)0, acc[i][j], false, false);
        __syncthreads();
    }

    const int mb = (lane >> 4) * 8;
    if (mode == 0) {
#pragma unroll
        for (int i = 0; i < 2; ++i)
#pragma unroll
            for (int j = 0; j < 2; ++j)
#pragma unroll
                for (int v = 0; v < 8; ++v) {
                    size_t o = (size_t)(bm + wm + i * 16 + mb + v) * N +
                               (bn + wn + j * 16 + ln15);
                    outF[o] = acc[i][j][v];
                }
    } else if (mode == 1) {
#pragma unroll
        for (int i = 0; i < 2; ++i)
#pragma unroll
            for (int j = 0; j < 2; ++j)
#pragma unroll
                for (int v = 0; v < 8; ++v) {
                    size_t o = (size_t)(bm + wm + i * 16 + mb + v) * N +
                               (bn + wn + j * 16 + ln15);
                    outF[o] = res[o] + acc[i][j][v];
                }
    } else {
#pragma unroll
        for (int i = 0; i < 2; ++i)
#pragma unroll
            for (int j = 0; j < 2; ++j)
#pragma unroll
                for (int v = 0; v < 8; ++v) {
                    size_t o = (size_t)(bm + wm + i * 16 + mb + v) * N +
                               (bn + wn + j * 16 + ln15);
                    outB[o] = to_bf16(acc[i][j][v]);
                }
    }
}

// ---------------- flash attention (sliding window, GQA) ----------------
// 1 wave per (batch, head, 16-query tile). q/k/v bf16 (B,L,heads,HD).
__global__ __launch_bounds__(32) void k_attention(
    const unsigned short* __restrict__ qb, const unsigned short* __restrict__ kb,
    const unsigned short* __restrict__ vb, unsigned short* __restrict__ ob) {
    __shared__ float        Ssh[16 * 32];
    __shared__ unsigned int Psh32[16 * 16];   // 16x32 bf16
    __shared__ unsigned int VT32[128 * 16];   // [hd][key] bf16, transposed
    __shared__ float        s_m[16], s_l[16], s_a[16];
    unsigned short* Psh16 = (unsigned short*)Psh32;
    unsigned short* VT16  = (unsigned short*)VT32;

    const int QT  = SEQL / 16;
    int gid = blockIdx.x;
    int qt  = gid % QT;
    int h   = (gid / QT) % NH;
    int b   = gid / (QT * NH);
    int q0  = qt * 16;
    int kvh = h / (NH / NKV);
    int lane = threadIdx.x;
    const int ln15 = lane & 15;
    const int mb   = (lane >> 4) * 8;

    // Q fragments (HD = 4 chunks of K=32) in registers
    Frag qf[4];
    {
        int koff = (lane >> 4) * 8;
        const unsigned short* qrow =
            qb + (((size_t)(b * SEQL + q0 + ln15)) * NH + h) * HD;
#pragma unroll
        for (int c = 0; c < 4; ++c)
#pragma unroll
            for (int j = 0; j < 4; ++j) {
                qf[c].u[j]     = *(const unsigned int*)(qrow + c * 32 + koff + 2 * j);
                qf[c].u[4 + j] = *(const unsigned int*)(qrow + c * 32 + 16 + koff + 2 * j);
            }
    }

    v8f o[8];
#pragma unroll
    for (int nc = 0; nc < 8; ++nc)
#pragma unroll
        for (int v = 0; v < 8; ++v) o[nc][v] = 0.f;
    if (lane < 16) { s_m[lane] = -1e30f; s_l[lane] = 0.f; }
    __syncthreads();

    int kstart = q0 - (WIN - 1);
    if (kstart < 0) kstart = 0;
    kstart &= ~31;
    const float SC = 11.313708498984761f;  // reference: scores / HD**(-0.5) = * sqrt(128)

    for (int kt = kstart; kt <= q0 + 15; kt += 32) {
        // stage V tile transposed: VT[hd][key]
        for (int e = lane; e < 32 * 64; e += 32) {
            int kk = e & 31, d2 = e >> 5;
            const unsigned short* vrow =
                vb + (((size_t)(b * SEQL + kt + kk)) * NKV + kvh) * HD;
            unsigned int w = *(const unsigned int*)(vrow + 2 * d2);
            VT16[(2 * d2) * 32 + kk]     = (unsigned short)(w & 0xFFFF);
            VT16[(2 * d2 + 1) * 32 + kk] = (unsigned short)(w >> 16);
        }
        // S = Q K^T  (two 16-key halves, scaled + masked in registers)
        for (int half = 0; half < 2; ++half) {
            v8f s;
#pragma unroll
            for (int v = 0; v < 8; ++v) s[v] = 0.f;
            int key  = kt + half * 16 + ln15;
            int kb16 = (lane >> 4) * 8;
            const unsigned short* krow =
                kb + (((size_t)(b * SEQL + key)) * NKV + kvh) * HD + kb16;
#pragma unroll
            for (int c = 0; c < 4; ++c) {
                Frag bfr;
#pragma unroll
                for (int j = 0; j < 8; ++j)
                    bfr.u[j] = *(const unsigned int*)(krow + c * 32 + (j >> 2) * 16 + 2 * (j & 3));
                s = __builtin_amdgcn_wmma_f32_16x16x32_bf16(
                    false, qf[c].v, false, bfr.v, (short)0, s, false, false);
            }
#pragma unroll
            for (int v = 0; v < 8; ++v) {
                int m = mb + v;
                int q = q0 + m;
                int k = kt + half * 16 + ln15;
                float val = s[v] * SC;
                if (k > q || (q - k) >= WIN) val = -1e30f;
                Ssh[m * 32 + half * 16 + ln15] = val;
            }
        }
        __syncthreads();
        // online softmax: lanes 0..15 own one query row each
        if (lane < 16) {
            int   m    = lane;
            float tmax = -1e30f;
            for (int n = 0; n < 32; ++n) tmax = fmaxf(tmax, Ssh[m * 32 + n]);
            float newm = fmaxf(s_m[m], tmax);
            float a    = __expf(s_m[m] - newm);
            float sum  = 0.f;
            for (int n = 0; n < 32; ++n) {
                float val = Ssh[m * 32 + n];
                float p   = (val <= -1e29f) ? 0.f : __expf(val - newm);
                sum += p;
                Psh16[m * 32 + n] = to_bf16(p);
            }
            s_l[m] = s_l[m] * a + sum;
            s_m[m] = newm;
            s_a[m] = a;
        }
        __syncthreads();
        // rescale O and accumulate P*V (P is one 16x32 A-fragment)
        Frag pf;
        {
            int k2 = (lane >> 4) * 4;
#pragma unroll
            for (int j = 0; j < 4; ++j) {
                pf.u[j]     = Psh32[ln15 * 16 + k2 + j];
                pf.u[4 + j] = Psh32[ln15 * 16 + 8 + k2 + j];
            }
        }
        float al[8];
#pragma unroll
        for (int v = 0; v < 8; ++v) al[v] = s_a[mb + v];
#pragma unroll
        for (int nc = 0; nc < 8; ++nc) {
#pragma unroll
            for (int v = 0; v < 8; ++v) o[nc][v] *= al[v];
            Frag vf;
            int hd  = nc * 16 + ln15;
            int kb2 = (lane >> 4) * 8;
#pragma unroll
            for (int j = 0; j < 8; ++j) vf.u[j] = VT32[hd * 16 + (j >> 2) * 8 + (lane >> 4) * 4 + (j & 3)];
            o[nc] = __builtin_amdgcn_wmma_f32_16x16x32_bf16(
                false, pf.v, false, vf.v, (short)0, o[nc], false, false);
        }
        __syncthreads();
    }

    float li[8];
#pragma unroll
    for (int v = 0; v < 8; ++v) li[v] = 1.f / s_l[mb + v];
#pragma unroll
    for (int nc = 0; nc < 8; ++nc)
#pragma unroll
        for (int v = 0; v < 8; ++v) {
            int m = mb + v;
            int n = nc * 16 + ln15;
            ob[(((size_t)(b * SEQL + q0 + m)) * NH + h) * HD + n] =
                to_bf16(o[nc][v] * li[v]);
        }
}

// ---------------- host-side orchestration ----------------
extern "C" void kernel_launch(void* const* d_in, const int* in_sizes, int n_in,
                              void* d_out, int out_size, void* d_ws, size_t ws_size,
                              hipStream_t stream) {
    (void)in_sizes; (void)n_in; (void)out_size; (void)ws_size;
    const int*   ids     = (const int*)d_in[0];
    const float* tok_emb = (const float*)d_in[1];
    const float* anw     = (const float*)d_in[2];
    const float* wq      = (const float*)d_in[3];
    const float* wk      = (const float*)d_in[4];
    const float* wv      = (const float*)d_in[5];
    const float* wo      = (const float*)d_in[6];
    const float* fnw     = (const float*)d_in[7];
    const float* w1      = (const float*)d_in[8];
    const float* w2      = (const float*)d_in[9];
    const float* w3      = (const float*)d_in[10];
    const float* nw      = (const float*)d_in[11];
    const float* outw    = (const float*)d_in[12];
    float*       out     = (float*)d_out;

    char* p = (char*)d_ws;
    auto alloc = [&](size_t bytes) -> char* {
        char* r = p;
        p += (bytes + 255) & ~(size_t)255;
        return r;
    };

    float*          hbuf  = (float*)alloc((size_t)BL * DIM * 4);
    unsigned short* xb    = (unsigned short*)alloc((size_t)BL * DIM * 2);
    float*          qf    = (float*)alloc((size_t)BL * NH * HD * 4);
    float*          kf    = (float*)alloc((size_t)BL * NKV * HD * 4);
    unsigned short* qbuf  = (unsigned short*)alloc((size_t)BL * NH * HD * 2);
    unsigned short* kbuf  = (unsigned short*)alloc((size_t)BL * NKV * HD * 2);
    unsigned short* vbuf  = (unsigned short*)alloc((size_t)BL * NKV * HD * 2);
    unsigned short* obuf  = (unsigned short*)alloc((size_t)BL * NH * HD * 2);
    float*          s1    = (float*)alloc((size_t)BL * FF * 4);
    float*          s2    = (float*)alloc((size_t)BL * FF * 4);
    unsigned short* gate  = (unsigned short*)alloc((size_t)BL * FF * 2);
    // all weights stored TRANSPOSED (N x K, bf16) for the GEMM B operand
    unsigned short* wqb   = (unsigned short*)alloc((size_t)NLAY * DIM * NH * HD * 2);
    unsigned short* wkb   = (unsigned short*)alloc((size_t)NLAY * DIM * NKV * HD * 2);
    unsigned short* wvb   = (unsigned short*)alloc((size_t)NLAY * DIM * NKV * HD * 2);
    unsigned short* wob   = (unsigned short*)alloc((size_t)NLAY * NH * HD * DIM * 2);
    unsigned short* w1b   = (unsigned short*)alloc((size_t)NLAY * DIM * FF * 2);
    unsigned short* w2b   = (unsigned short*)alloc((size_t)NLAY * FF * DIM * 2);
    unsigned short* w3b   = (unsigned short*)alloc((size_t)NLAY * DIM * FF * 2);
    unsigned short* outwb = (unsigned short*)alloc((size_t)DIM * VOCAB * 2);

    auto cvtT = [&](const float* s, unsigned short* d, int K, int N) {
        dim3 g(N / 32, K / 32);
        k_transpose_bf16<<<g, 256, 0, stream>>>(s, d, K, N);
    };
    auto gemm = [&](const unsigned short* A, const unsigned short* Bt, float* oF,
                    unsigned short* oB, const float* res, int M, int N, int K,
                    int mode) {
        dim3 g(N / 64, M / 128);
        k_gemm_bf16<<<g, 256, 0, stream>>>(A, Bt, oF, oB, res, M, N, K, mode);
    };

    // weight conversions: f32 (K x N) -> bf16 transposed (N x K); re-done every
    // launch for determinism
    for (int l = 0; l < NLAY; ++l) {
        cvtT(wq + (size_t)l * DIM * NH * HD,  wqb + (size_t)l * DIM * NH * HD,  DIM, NH * HD);
        cvtT(wk + (size_t)l * DIM * NKV * HD, wkb + (size_t)l * DIM * NKV * HD, DIM, NKV * HD);
        cvtT(wv + (size_t)l * DIM * NKV * HD, wvb + (size_t)l * DIM * NKV * HD, DIM, NKV * HD);
        cvtT(wo + (size_t)l * NH * HD * DIM,  wob + (size_t)l * NH * HD * DIM,  NH * HD, DIM);
        cvtT(w1 + (size_t)l * DIM * FF,       w1b + (size_t)l * DIM * FF,       DIM, FF);
        cvtT(w2 + (size_t)l * FF * DIM,       w2b + (size_t)l * FF * DIM,       FF, DIM);
        cvtT(w3 + (size_t)l * DIM * FF,       w3b + (size_t)l * DIM * FF,       DIM, FF);
    }
    cvtT(outw, outwb, DIM, VOCAB);

    k_embed<<<2048, 256, 0, stream>>>(ids, tok_emb, hbuf);

    for (int l = 0; l < NLAY; ++l) {
        // attention block
        k_rmsnorm<<<BL, 256, 0, stream>>>(hbuf, anw + (size_t)l * DIM, xb);
        gemm(xb, wqb + (size_t)l * DIM * NH * HD,  qf, nullptr, nullptr, BL, NH * HD,  DIM, 0);
        gemm(xb, wkb + (size_t)l * DIM * NKV * HD, kf, nullptr, nullptr, BL, NKV * HD, DIM, 0);
        gemm(xb, wvb + (size_t)l * DIM * NKV * HD, nullptr, vbuf, nullptr, BL, NKV * HD, DIM, 2);
        k_rope<<<2048, 256, 0, stream>>>(qf, qbuf, NH);
        k_rope<<<2048, 256, 0, stream>>>(kf, kbuf, NKV);
        k_attention<<<BATCH * NH * (SEQL / 16), 32, 0, stream>>>(qbuf, kbuf, vbuf, obuf);
        gemm(obuf, wob + (size_t)l * NH * HD * DIM, hbuf, nullptr, hbuf, BL, DIM, NH * HD, 1);
        // FFN block
        k_rmsnorm<<<BL, 256, 0, stream>>>(hbuf, fnw + (size_t)l * DIM, xb);
        gemm(xb, w1b + (size_t)l * DIM * FF, s1, nullptr, nullptr, BL, FF, DIM, 0);
        gemm(xb, w3b + (size_t)l * DIM * FF, s2, nullptr, nullptr, BL, FF, DIM, 0);
        k_silu_mul<<<2048, 256, 0, stream>>>(s1, s2, gate, (size_t)BL * FF);
        gemm(gate, w2b + (size_t)l * FF * DIM, hbuf, nullptr, hbuf, BL, DIM, FF, 1);
    }

    // final norm + LM head
    k_rmsnorm<<<BL, 256, 0, stream>>>(hbuf, nw, xb);
    gemm(xb, outwb, out, nullptr, nullptr, BL, VOCAB, DIM, 0);
}
